// GatedGraphConv_28080496181509
// MI455X (gfx1250) — matrix operand order repaired and checked by reading, not compiled, not run on hardware
//
#include <hip/hip_runtime.h>
#include <stdint.h>

#define C_DIM   512
#define KDIM    512
#define MAXDEG  16
#define NLAYERS 4

#define BM      128
#define BN      64
#define KSTEP   32
#define BSTRIDE (KDIM + 8)   // bf16 elements per LDS row; pad keeps 16B align, spreads banks

typedef __attribute__((ext_vector_type(16))) __bf16       v16bf;
typedef __attribute__((ext_vector_type(8)))  float        v8f;
typedef __attribute__((ext_vector_type(4)))  unsigned int u32x4;
typedef __attribute__((ext_vector_type(2)))  unsigned int u32x2;

union FragAB { v16bf v; u32x4 q[2]; };
union Acc    { v8f v; float f[8]; };

__device__ __forceinline__ uint16_t f32_to_bf16(float f) {
  uint32_t u = __float_as_uint(f);
  u += 0x7FFFu + ((u >> 16) & 1u);      // round-to-nearest-even
  return (uint16_t)(u >> 16);
}
__device__ __forceinline__ float bf16_to_f32(uint32_t h) {
  return __uint_as_float(h << 16);
}

// ---------------- conversion kernels ----------------

__global__ __launch_bounds__(256) void convert_kernel(
    const float* __restrict__ in, uint16_t* __restrict__ out, int total) {
  int t = blockIdx.x * blockDim.x + threadIdx.x;
  if (t < total) out[t] = f32_to_bf16(in[t]);
}

// out[l][j][k] = in[l][k][j], C_DIM = 512 (2^9), C*C = 2^18
__global__ __launch_bounds__(256) void transpose_convert_kernel(
    const float* __restrict__ in, uint16_t* __restrict__ out, int total) {
  int t = blockIdx.x * blockDim.x + threadIdx.x;
  if (t >= total) return;
  int l   = t >> 18;
  int rem = t & 262143;
  int j   = rem >> 9;
  int k   = rem & 511;
  out[t] = f32_to_bf16(in[(l << 18) + (k << 9) + j]);
}

// ---------------- bf16 NT GEMM: Out = A[MxK] * Bt[NcolsxK]^T (+bias), K = 512 ----------------
// B panel (64 cols x 512 K, 64KB bf16) is staged ONCE into LDS via CDNA5 async
// global->LDS DMA (ASYNCcnt), then the K-loop runs barrier-free.

__global__ __launch_bounds__(256) void gemm_bf16_nt(
    const uint16_t* __restrict__ A,    // [M][512] bf16 row-major
    const uint16_t* __restrict__ Bt,   // [Ncols][512] bf16 row-major (B transposed)
    const float* __restrict__ bias,    // [Ncols] or nullptr
    float* __restrict__ outF,          // [M][Ncols] f32 or nullptr
    uint16_t* __restrict__ outB,       // [M][Ncols] bf16 or nullptr
    int M, int Ncols)
{
  __shared__ __align__(16) uint16_t bsh[BN * BSTRIDE];   // 64 x 520 bf16 = 66,560 B

  const int tid  = threadIdx.x;
  const int lane = tid & 31;
  const int wid  = tid >> 5;
  const int wm   = wid & 3;      // 4 waves along M
  const int wn   = wid >> 2;     // 2 waves along N
  const int hl   = lane >> 4;    // half-wave select
  const int l16  = lane & 15;
  const int rowBlk = blockIdx.x * BM;
  const int colBlk = blockIdx.y * BN;

  // ---- prologue: async-DMA the 64x512 B panel into LDS (16 x b128 per thread) ----
  {
    const int r   = tid >> 2;            // 0..63: panel row (output column)
    const int seg = (tid & 3) * 128;     // 4 threads cover one 512-elem row
    const uint16_t* gsrc = Bt + (long)(colBlk + r) * KDIM + seg;
    uint32_t ldsoff = (uint32_t)(uintptr_t)(&bsh[r * BSTRIDE + seg]);
#pragma unroll
    for (int i = 0; i < 16; ++i) {
      uint64_t ga = (uint64_t)(uintptr_t)(gsrc + i * 8);
      uint32_t lo = ldsoff + i * 16;
      asm volatile("global_load_async_to_lds_b128 %0, %1, off"
                   :: "v"(lo), "v"(ga) : "memory");
    }
    asm volatile("s_wait_asynccnt 0x0" ::: "memory");
  }
  __syncthreads();

  Acc acc[2][2];
#pragma unroll
  for (int i = 0; i < 2; ++i)
#pragma unroll
    for (int j = 0; j < 2; ++j)
#pragma unroll
      for (int v = 0; v < 8; ++v) acc[i][j].f[v] = 0.0f;

  long abase[2];
#pragma unroll
  for (int mt = 0; mt < 2; ++mt) {
    int r = rowBlk + wm * 32 + mt * 16 + l16;
    if (r >= M) r = M - 1;               // clamp; tail rows discarded at store
    abase[mt] = (long)r * KDIM;
  }

  // ---- barrier-free K loop ----
#pragma unroll 4
  for (int kb = 0; kb < KDIM; kb += KSTEP) {
    FragAB a[2], b[2];
    // A fragment: lane holds row l16, K = {kb+hl*8..+7} U {kb+16+hl*8..+7}
#pragma unroll
    for (int mt = 0; mt < 2; ++mt) {
      const uint16_t* p = A + abase[mt] + kb + hl * 8;
      a[mt].q[0] = *reinterpret_cast<const u32x4*>(p);
      a[mt].q[1] = *reinterpret_cast<const u32x4*>(p + 16);
    }
    // B fragment: lane holds col l16, K = {kb+hl*8..+7} U {kb+16+hl*8..+7}
#pragma unroll
    for (int nt = 0; nt < 2; ++nt) {
      const uint16_t* p = &bsh[(wn * 32 + nt * 16 + l16) * BSTRIDE + kb + hl * 8];
      b[nt].q[0] = *reinterpret_cast<const u32x4*>(p);
      b[nt].q[1] = *reinterpret_cast<const u32x4*>(p + 16);
    }
#pragma unroll
    for (int mt = 0; mt < 2; ++mt)
#pragma unroll
      for (int nt = 0; nt < 2; ++nt)
        acc[mt][nt].v = __builtin_amdgcn_wmma_f32_16x16x32_bf16(
            false, a[mt].v, false, b[nt].v, (short)0, acc[mt][nt].v, false, false);
  }

  // epilogue: C/D layout -> row = base + hl*8 + v, col = base + l16
#pragma unroll
  for (int mt = 0; mt < 2; ++mt) {
#pragma unroll
    for (int nt = 0; nt < 2; ++nt) {
      const int col = colBlk + wn * 32 + nt * 16 + l16;
      const float bv = bias ? bias[col] : 0.0f;
#pragma unroll
      for (int v = 0; v < 8; ++v) {
        const int row = rowBlk + wm * 32 + mt * 16 + hl * 8 + v;
        if (row < M) {
          const float val = acc[mt][nt].f[v] + bv;
          if (outF) outF[(long)row * Ncols + col] = val;
          if (outB) outB[(long)row * Ncols + col] = f32_to_bf16(val);
        }
      }
    }
  }
}

// ---------------- neighbor gather-sum (m is bf16, L2-resident) ----------------

__global__ __launch_bounds__(128) void gather_kernel(
    const uint16_t* __restrict__ m, const int* __restrict__ edges,
    uint16_t* __restrict__ msg, int nnodes)
{
  const int n = blockIdx.x;
  const int c = threadIdx.x * 4;         // 128 threads x 4 cols = 512
  int idxbuf[MAXDEG];
#pragma unroll
  for (int d = 0; d < MAXDEG; ++d) idxbuf[d] = edges[(long)n * MAXDEG + d];

  float a0 = 0.f, a1 = 0.f, a2 = 0.f, a3 = 0.f;
#pragma unroll
  for (int d = 0; d < MAXDEG; ++d) {
    if (d + 8 < MAXDEG) {
      int pi = idxbuf[d + 8];
      if ((unsigned)pi < (unsigned)nnodes)
        __builtin_prefetch(m + (long)pi * C_DIM + c, 0, 0);   // global_prefetch_b8
    }
    const int idx = idxbuf[d];
    if ((unsigned)idx < (unsigned)nnodes) {
      const u32x2 pv = *reinterpret_cast<const u32x2*>(m + (long)idx * C_DIM + c);
      a0 += bf16_to_f32(pv.x & 0xFFFFu);
      a1 += bf16_to_f32(pv.x >> 16);
      a2 += bf16_to_f32(pv.y & 0xFFFFu);
      a3 += bf16_to_f32(pv.y >> 16);
    }
  }
  u32x2 o;
  o.x = (uint32_t)f32_to_bf16(a0) | ((uint32_t)f32_to_bf16(a1) << 16);
  o.y = (uint32_t)f32_to_bf16(a2) | ((uint32_t)f32_to_bf16(a3) << 16);
  *reinterpret_cast<u32x2*>(msg + (long)n * C_DIM + c) = o;
}

// ---------------- fused GRU elementwise ----------------

__global__ __launch_bounds__(256) void gru_kernel(
    const float* __restrict__ gi, const float* __restrict__ gh,
    const float* __restrict__ hin, float* __restrict__ hout,
    uint16_t* __restrict__ hout_bf, int total)
{
  int t = blockIdx.x * blockDim.x + threadIdx.x;
  if (t >= total) return;
  const int n = t >> 9;                   // /512
  const int c = t & 511;
  const long base = (long)n * (3 * C_DIM) + c;
  const float ir = gi[base], iz = gi[base + C_DIM], in_ = gi[base + 2 * C_DIM];
  const float hr = gh[base], hz = gh[base + C_DIM], hn  = gh[base + 2 * C_DIM];
  const float r  = 1.0f / (1.0f + expf(-(ir + hr)));
  const float z  = 1.0f / (1.0f + expf(-(iz + hz)));
  const float nn = tanhf(in_ + r * hn);
  const float h  = hin[t];
  const float o  = (1.0f - z) * nn + z * h;
  hout[t] = o;
  if (hout_bf) hout_bf[t] = f32_to_bf16(o);
}

// ---------------- driver ----------------

extern "C" void kernel_launch(void* const* d_in, const int* in_sizes, int n_in,
                              void* d_out, int out_size, void* d_ws, size_t ws_size,
                              hipStream_t stream) {
  (void)n_in; (void)out_size; (void)ws_size;
  const float* x    = (const float*)d_in[0];
  const int*   edge = (const int*)d_in[1];
  const float* wgt  = (const float*)d_in[2];
  const float* w_ih = (const float*)d_in[3];
  const float* w_hh = (const float*)d_in[4];
  const float* b_ih = (const float*)d_in[5];
  const float* b_hh = (const float*)d_in[6];
  float* out = (float*)d_out;

  const int N = in_sizes[0] / C_DIM;      // 50000

  char* p = (char*)d_ws;
  auto take = [&](size_t bytes) -> char* {
    char* r = p; p += (bytes + 255) & ~(size_t)255; return r;
  };
  float*    h_f32  = (float*)   take((size_t)N * C_DIM * 4);
  uint16_t* h_bf   = (uint16_t*)take((size_t)N * C_DIM * 2);
  uint16_t* m_bf   = (uint16_t*)take((size_t)N * C_DIM * 2);
  uint16_t* msg_bf = (uint16_t*)take((size_t)N * C_DIM * 2);
  float*    gi     = (float*)   take((size_t)N * 3 * C_DIM * 4);
  float*    gh     = (float*)   take((size_t)N * 3 * C_DIM * 4);
  uint16_t* wt_bf  = (uint16_t*)take((size_t)NLAYERS * C_DIM * C_DIM * 2);
  uint16_t* wih_bf = (uint16_t*)take((size_t)3 * C_DIM * C_DIM * 2);
  uint16_t* whh_bf = (uint16_t*)take((size_t)3 * C_DIM * C_DIM * 2);

  // per-call setup: convert weights (transpose per-layer weight), hidden state
  {
    int total = NLAYERS * C_DIM * C_DIM;
    transpose_convert_kernel<<<(total + 255) / 256, 256, 0, stream>>>(wgt, wt_bf, total);
  }
  {
    int total = 3 * C_DIM * C_DIM;
    convert_kernel<<<(total + 255) / 256, 256, 0, stream>>>(w_ih, wih_bf, total);
    convert_kernel<<<(total + 255) / 256, 256, 0, stream>>>(w_hh, whh_bf, total);
  }
  {
    int total = N * C_DIM;
    convert_kernel<<<(total + 255) / 256, 256, 0, stream>>>(x, h_bf, total);
  }

  const dim3 blk(256);
  const dim3 gridC((N + BM - 1) / BM, C_DIM / BN);
  const dim3 grid3C((N + BM - 1) / BM, (3 * C_DIM) / BN);
  const int totalNC = N * C_DIM;

  for (int layer = 0; layer < NLAYERS; ++layer) {
    // m = h @ W[layer]   (bf16 out, stays L2-resident for the gather)
    gemm_bf16_nt<<<gridC, blk, 0, stream>>>(
        h_bf, wt_bf + (size_t)layer * C_DIM * C_DIM,
        nullptr, nullptr, m_bf, N, C_DIM);
    // msg[n] = sum_d m[edge[n][d]]
    gather_kernel<<<N, 128, 0, stream>>>(m_bf, edge, msg_bf, N);
    // gi = msg @ w_ih^T + b_ih ; gh = h @ w_hh^T + b_hh
    gemm_bf16_nt<<<grid3C, blk, 0, stream>>>(
        msg_bf, wih_bf, b_ih, gi, nullptr, N, 3 * C_DIM);
    gemm_bf16_nt<<<grid3C, blk, 0, stream>>>(
        h_bf, whh_bf, b_hh, gh, nullptr, N, 3 * C_DIM);
    // GRU update (writes next-layer f32 + bf16 hidden; last layer -> d_out)
    const float* hin  = (layer == 0) ? x : h_f32;
    float*    hout    = (layer == NLAYERS - 1) ? out : h_f32;
    uint16_t* hboutbf = (layer == NLAYERS - 1) ? nullptr : h_bf;
    gru_kernel<<<(totalNC + 255) / 256, 256, 0, stream>>>(
        gi, gh, hin, hout, hboutbf, totalNC);
  }
}
